// WorldModel_3882650436754
// MI455X (gfx1250) — compile-verified, matrix operand order checked by reading, not run on hardware
//
#include <hip/hip_runtime.h>

#define B_   512
#define T_   64
#define OBS_ 64
#define ACT_ 6
#define S_   30
#define H_   200
#define E_   1024
#define BT_  (B_*T_)

typedef __bf16 bf16_t;
typedef __attribute__((ext_vector_type(16))) __bf16 v16bf;
typedef __attribute__((ext_vector_type(8)))  float  v8f;

typedef union { uint4 u[2]; v16bf v; } frag_t;

__device__ __forceinline__ float sigmoidf_(float x){ return 1.0f/(1.0f + __expf(-x)); }

// ---------------------------------------------------------------------------
// WMMA GEMM:  C = A(MxK) * W(NxK)^T + bias, bf16 inputs, f32 accumulation.
// A: row-major MxK (lda >= K, K contiguous).  W: row-major NxK (ldw >= K).
// Each wave computes a 16 x (NW*16) strip (one A fragment reused for NW
// WMMAs).  Explicit 2-stage software pipeline with two named fragment sets:
// the next K-step's loads are issued before the current step's WMMAs and are
// data-independent of them, forcing load/WMMA overlap instead of a full
// s_wait_loadcnt 0 drain per WMMA.  Requires K >= 64, K % 32 == 0 (all true).
// ACTMODE: 0 = none, 1 = relu, 2 = sigmoid.
// ---------------------------------------------------------------------------
template<int ACTMODE, int NW>
__global__ __launch_bounds__(256) void gemm_wmma_bf16(
    const bf16_t* __restrict__ A, int lda,
    const bf16_t* __restrict__ W, int ldw,
    const float*  __restrict__ bias,
    float*        __restrict__ Cf, int ldcf, int Nreal,
    bf16_t*       __restrict__ Cb, int ldcb,
    int M, int Npad, int K)
{
  const int lane    = threadIdx.x & 31;
  const int wave    = threadIdx.x >> 5;
  const int nstrips = (Npad >> 4) / NW;
  const int strip   = blockIdx.x * 8 + wave;
  if (strip >= (M >> 4) * nstrips) return;
  const int mt   = strip / nstrips;
  const int ns   = strip % nstrips;        // covers n-tiles ns*NW .. ns*NW+NW-1
  const int half = lane >> 4;
  const int lr   = lane & 15;

  // 16-bit A-matrix 16x32 layout (ISA 7.12.2): lanes 0-15 hold K 0..7,16..23;
  // lanes 16-31 hold K 8..15,24..31.  B (W row-major over K) mirrors it.
  const bf16_t* pA = A + (size_t)(mt*16 + lr) * lda + half*8;
  const bf16_t* pW = W + (size_t)(ns*NW*16 + lr) * ldw + half*8;

  v8f acc[NW];
  const v8f vzero = {0.f,0.f,0.f,0.f,0.f,0.f,0.f,0.f};
  #pragma unroll
  for (int i = 0; i < NW; ++i) acc[i] = vzero;

  auto load_set = [&](frag_t& fa, frag_t* fb, int kk){
    const bf16_t* a = pA + kk;
    fa.u[0] = *(const uint4*)(a);
    fa.u[1] = *(const uint4*)(a + 16);
    #pragma unroll
    for (int i = 0; i < NW; ++i) {
      const bf16_t* w = pW + (size_t)i * 16 * ldw + kk;
      fb[i].u[0] = *(const uint4*)(w);
      fb[i].u[1] = *(const uint4*)(w + 16);
    }
  };
  auto mma_set = [&](const frag_t& fa, const frag_t* fb){
    #pragma unroll
    for (int i = 0; i < NW; ++i)
      acc[i] = __builtin_amdgcn_wmma_f32_16x16x32_bf16(false, fa.v, false, fb[i].v,
                                                       (short)0, acc[i], false, false);
  };

  frag_t fa0, fa1, fb0[NW], fb1[NW];
  load_set(fa0, fb0, 0);                    // prologue: step 0 in flight
  int k = 32;
  for (; k + 64 <= K; k += 64) {            // steady state: 2 steps/iter
    load_set(fa1, fb1, k);                  // step k    -> set1 (independent)
    mma_set(fa0, fb0);                      // step k-32 -> XDL
    __builtin_prefetch(pA + k + 256, 0, 3); // near prefetch of A stream
    load_set(fa0, fb0, k + 32);             // step k+32 -> set0 (independent)
    mma_set(fa1, fb1);                      // step k    -> XDL
  }
  if (k < K) {                              // K % 64 == 0: one step remains
    load_set(fa1, fb1, k);
    mma_set(fa0, fb0);
    mma_set(fa1, fb1);
  } else {                                  // K % 64 == 32: all loaded already
    mma_set(fa0, fb0);
  }

  // C/D layout: lanes 0-15 -> (M = r,   N = lane),
  //             lanes 16-31 -> (M = 8+r, N = lane-16)
  #pragma unroll
  for (int i = 0; i < NW; ++i) {
    const int   n   = (ns*NW + i)*16 + lr;
    const float bv  = bias ? bias[n] : 0.f;
    const bool  okn = (n < Nreal);
    #pragma unroll
    for (int r = 0; r < 8; ++r) {
      float v = acc[i][r] + bv;
      if (ACTMODE == 1) v = fmaxf(v, 0.f);
      if (ACTMODE == 2) v = sigmoidf_(v);
      const int m = mt*16 + half*8 + r;
      if (Cf && okn) Cf[(size_t)m * ldcf + n] = v;
      if (Cb)        Cb[(size_t)m * ldcb + n] = okn ? (bf16_t)v : (bf16_t)0.f;
    }
  }
}

// ---------------------------------------------------------------------------
// Packing / elementwise kernels
// ---------------------------------------------------------------------------
__global__ void k_pack_w(const float* __restrict__ w, int N, int K,
                         bf16_t* __restrict__ out, int Np, int Kp) {
  int i = blockIdx.x*blockDim.x + threadIdx.x;
  if (i >= Np*Kp) return;
  int n = i / Kp, k = i % Kp;
  float v = (n < N && k < K) ? w[(size_t)n*K + k] : 0.f;
  out[i] = (bf16_t)v;
}

__global__ void k_pack_bias(const float* __restrict__ b, int N,
                            float* __restrict__ out, int Np){
  int i = blockIdx.x*blockDim.x + threadIdx.x;
  if (i < Np) out[i] = (i < N) ? b[i] : 0.f;
}

__global__ void k_cast_bf16(const float* __restrict__ in, bf16_t* __restrict__ out, size_t n){
  size_t i = (size_t)blockIdx.x*blockDim.x + threadIdx.x;
  if (i < n) out[i] = (bf16_t)in[i];
}

__global__ void k_zero_f32(float* p, size_t n){
  size_t i = (size_t)blockIdx.x*blockDim.x + threadIdx.x;
  if (i < n) p[i] = 0.f;
}

__global__ void k_zero_bf16(bf16_t* p, size_t n){
  size_t i = (size_t)blockIdx.x*blockDim.x + threadIdx.x;
  if (i < n) p[i] = (bf16_t)0.f;
}

// x = [s (30), a_{t-1} (6), pad] as bf16, 512 x 64
__global__ void k_pack_x(const float* __restrict__ s, const float* __restrict__ act, int t,
                         bf16_t* __restrict__ x){
  int i = blockIdx.x*blockDim.x + threadIdx.x;
  if (i >= B_*64) return;
  int b = i >> 6, k = i & 63;
  float v = 0.f;
  if (k < S_)                       v = s[(size_t)b*S_ + k];
  else if (k < S_ + ACT_ && t > 0)  v = act[((size_t)b*T_ + (t-1))*ACT_ + (k - S_)];
  x[i] = (bf16_t)v;
}

// GRU gates: h_new = (1-z)*n + z*h ; writes h (f32, in place), h_bf (padded),
// and the hidden part of the decoder input slab sc[b*T+t, 30..229].
__global__ void k_gate(const float* __restrict__ gx, const float* __restrict__ gh,
                       float* __restrict__ h, bf16_t* __restrict__ h_bf,
                       bf16_t* __restrict__ sc, int t){
  int i = blockIdx.x*blockDim.x + threadIdx.x;
  if (i >= B_*224) return;
  int b = i / 224, j = i % 224;
  if (j >= H_) { h_bf[i] = (bf16_t)0.f; return; }
  size_t o = (size_t)b*600 + j;
  float r = sigmoidf_(gx[o]        + gh[o]);
  float z = sigmoidf_(gx[o + H_]   + gh[o + H_]);
  float n = tanhf(    gx[o + 2*H_] + r*gh[o + 2*H_]);
  float hp = h[(size_t)b*H_ + j];
  float hn = (1.f - z)*n + z*hp;
  h[(size_t)b*H_ + j] = hn;
  h_bf[i] = (bf16_t)hn;
  sc[((size_t)b*T_ + t)*256 + S_ + j] = (bf16_t)hn;
}

// xpos = [h_new (200), emb[:,t,:] (1024), pad] as bf16, 512 x 1248
__global__ void k_pack_xpos(const float* __restrict__ h, const bf16_t* __restrict__ emb, int t,
                            bf16_t* __restrict__ xp){
  int i = blockIdx.x*blockDim.x + threadIdx.x;
  if (i >= B_*1248) return;
  int b = i / 1248, k = i % 1248;
  bf16_t v;
  if (k < H_)           v = (bf16_t)h[(size_t)b*H_ + k];
  else if (k < H_ + E_) v = emb[((size_t)b*T_ + t)*E_ + (k - H_)];
  else                  v = (bf16_t)0.f;
  xp[i] = v;
}

// sample s = pm + exp(pls)*nz, write s (f32 + sc slab) and KL directly to out
__global__ void k_skl(const float* __restrict__ post, const float* __restrict__ pri,
                      const float* __restrict__ noise, int t,
                      float* __restrict__ s, bf16_t* __restrict__ sc,
                      float* __restrict__ kl){
  int b = blockIdx.x*blockDim.x + threadIdx.x;
  if (b >= B_) return;
  float acc = 0.f;
  for (int j = 0; j < S_; ++j) {
    float pm   = post[(size_t)b*2*S_ + j];
    float pls  = post[(size_t)b*2*S_ + S_ + j];
    float pstd = __expf(pls);
    float nz   = noise[((size_t)b*T_ + t)*S_ + j];
    float sv   = pm + pstd*nz;
    s[(size_t)b*S_ + j] = sv;
    sc[((size_t)b*T_ + t)*256 + j] = (bf16_t)sv;
    float prm   = pri[(size_t)b*2*S_ + j];
    float prls  = pri[(size_t)b*2*S_ + S_ + j];
    float prstd = __expf(prls);
    float v1 = pstd*pstd, v2 = prstd*prstd;
    float d  = prm - pm;
    acc += v1/v2 + d*d/v2 + (prls - pls) - 1.f;
  }
  kl[(size_t)b*T_ + t] = 0.5f*acc;
}

// ---------------------------------------------------------------------------
extern "C" void kernel_launch(void* const* d_in, const int* in_sizes, int n_in,
                              void* d_out, int out_size, void* d_ws, size_t ws_size,
                              hipStream_t stream) {
  (void)in_sizes; (void)n_in; (void)out_size; (void)ws_size;

  const float* obs_seq = (const float*)d_in[0];
  const float* act_seq = (const float*)d_in[1];
  const float* noise   = (const float*)d_in[2];

  char*  base = (char*)d_ws;
  size_t off  = 0;
  auto alloc = [&](size_t bytes)->void* {
    off = (off + 255) & ~(size_t)255;
    void* p = base + off;
    off += bytes;
    return p;
  };

  // ---- weight packing (f32 -> padded bf16; biases -> padded f32) ----------
  struct WS { int wi, bi, N, K, Np, Kp; };
  const WS wspec[18] = {
    {3,4,   256, OBS_,   256,  64  },  // enc1
    {5,6,   512, 256,    512,  256 },  // enc2
    {7,8,   E_,  512,    E_,   512 },  // enc3
    {9,11,  600, S_+ACT_,608,  64  },  // gru ih (bias = bih)
    {10,12, 600, H_,     608,  224 },  // gru hh (bias = bhh)
    {13,14, H_,  H_,     224,  224 },  // pri1
    {15,16, 2*S_,H_,     64,   224 },  // pri2
    {17,18, H_,  H_+E_,  224,  1248},  // pos1
    {19,20, 2*S_,H_,     64,   224 },  // pos2
    {21,22, 512, S_+H_,  512,  256 },  // dec1
    {23,24, 256, 512,    256,  512 },  // dec2
    {25,26, OBS_,256,    64,   256 },  // dec3
    {27,28, 256, S_+H_,  256,  256 },  // rew1
    {29,30, 128, 256,    128,  256 },  // rew2
    {31,32, 1,   128,    16,   128 },  // rew3
    {33,34, 128, S_+H_,  128,  256 },  // con1
    {35,36, 64,  128,    64,   128 },  // con2
    {37,38, 1,   64,     16,   64  },  // con3
  };
  enum { W_ENC1=0, W_ENC2, W_ENC3, W_GIH, W_GHH, W_PRI1, W_PRI2, W_POS1, W_POS2,
         W_DEC1, W_DEC2, W_DEC3, W_REW1, W_REW2, W_REW3, W_CON1, W_CON2, W_CON3 };

  bf16_t* Wb[18]; float* Bb[18];
  for (int i = 0; i < 18; ++i) {
    const WS& s = wspec[i];
    Wb[i] = (bf16_t*)alloc((size_t)s.Np * s.Kp * sizeof(bf16_t));
    Bb[i] = (float*) alloc((size_t)s.Np * sizeof(float));
    int tot = s.Np * s.Kp;
    k_pack_w   <<<(tot + 255)/256, 256, 0, stream>>>((const float*)d_in[s.wi], s.N, s.K, Wb[i], s.Np, s.Kp);
    k_pack_bias<<<(s.Np + 255)/256, 256, 0, stream>>>((const float*)d_in[s.bi], s.N, Bb[i], s.Np);
  }

  // ---- activations --------------------------------------------------------
  bf16_t* obs_bf  = (bf16_t*)alloc((size_t)BT_*OBS_*sizeof(bf16_t));
  bf16_t* emb_bf  = (bf16_t*)alloc((size_t)BT_*E_  *sizeof(bf16_t));
  bf16_t* sc_bf   = (bf16_t*)alloc((size_t)BT_*256 *sizeof(bf16_t));
  bf16_t* bufA    = (bf16_t*)alloc((size_t)BT_*512 *sizeof(bf16_t));
  bf16_t* bufB    = (bf16_t*)alloc((size_t)BT_*256 *sizeof(bf16_t));
  bf16_t* x_bf    = (bf16_t*)alloc((size_t)B_*64   *sizeof(bf16_t));
  bf16_t* h_bf    = (bf16_t*)alloc((size_t)B_*224  *sizeof(bf16_t));
  bf16_t* xpos_bf = (bf16_t*)alloc((size_t)B_*1248 *sizeof(bf16_t));
  bf16_t* p1_bf   = (bf16_t*)alloc((size_t)B_*224  *sizeof(bf16_t));
  bf16_t* q1_bf   = (bf16_t*)alloc((size_t)B_*224  *sizeof(bf16_t));
  float*  h_f32   = (float*) alloc((size_t)B_*H_  *sizeof(float));
  float*  s_f32   = (float*) alloc((size_t)B_*S_  *sizeof(float));
  float*  gx_f    = (float*) alloc((size_t)B_*600 *sizeof(float));
  float*  gh_f    = (float*) alloc((size_t)B_*600 *sizeof(float));
  float*  post_f  = (float*) alloc((size_t)B_*2*S_*sizeof(float));
  float*  pri_f   = (float*) alloc((size_t)B_*2*S_*sizeof(float));

  float* out       = (float*)d_out;
  float* recon_out = out;                          // (B,T,OBS)
  float* rew_out   = out + (size_t)BT_*OBS_;       // (B,T,1)
  float* con_out   = rew_out + BT_;                // (B,T,1)
  float* kl_out    = con_out + BT_;                // (B,T,1)

  auto gemm = [&](int act, const bf16_t* A, int lda, const bf16_t* W, int ldw,
                  const float* bias, float* Cf, int ldcf, int Nreal,
                  bf16_t* Cb, int ldcb, int M, int Npad, int K){
    const int ntiles = Npad / 16;
    const int nw     = (ntiles % 4 == 0) ? 4 : (ntiles % 2 == 0) ? 2 : 1;
    const int strips = (M/16) * (ntiles/nw);
    const int blocks = (strips + 7) / 8;
    #define GEMM_CASE(AM, NWv) \
      gemm_wmma_bf16<AM,NWv><<<blocks,256,0,stream>>>(A,lda,W,ldw,bias,Cf,ldcf,Nreal,Cb,ldcb,M,Npad,K)
    if (act == 0)      { if (nw==4) GEMM_CASE(0,4); else if (nw==2) GEMM_CASE(0,2); else GEMM_CASE(0,1); }
    else if (act == 1) { if (nw==4) GEMM_CASE(1,4); else if (nw==2) GEMM_CASE(1,2); else GEMM_CASE(1,1); }
    else               { if (nw==4) GEMM_CASE(2,4); else if (nw==2) GEMM_CASE(2,2); else GEMM_CASE(2,1); }
    #undef GEMM_CASE
  };

  // ---- init recurrent state + sc padding ----------------------------------
  k_zero_f32 <<<((size_t)B_*H_  + 255)/256, 256, 0, stream>>>(h_f32, (size_t)B_*H_);
  k_zero_f32 <<<((size_t)B_*S_  + 255)/256, 256, 0, stream>>>(s_f32, (size_t)B_*S_);
  k_zero_bf16<<<((size_t)B_*224 + 255)/256, 256, 0, stream>>>(h_bf,  (size_t)B_*224);
  k_zero_bf16<<<((size_t)BT_*256+ 255)/256, 256, 0, stream>>>(sc_bf, (size_t)BT_*256);

  // ---- encoder: obs -> 256 -> 512 -> E (relu, relu, linear) ---------------
  k_cast_bf16<<<((size_t)BT_*OBS_ + 255)/256, 256, 0, stream>>>(obs_seq, obs_bf, (size_t)BT_*OBS_);
  gemm(1, obs_bf, 64,  Wb[W_ENC1], 64,  Bb[W_ENC1], nullptr,0,256,  bufB,  256,  BT_, 256,  64 );
  gemm(1, bufB,   256, Wb[W_ENC2], 256, Bb[W_ENC2], nullptr,0,512,  bufA,  512,  BT_, 512,  256);
  gemm(0, bufA,   512, Wb[W_ENC3], 512, Bb[W_ENC3], nullptr,0,E_,   emb_bf,E_,   BT_, E_,   512);

  // ---- GRU scan -----------------------------------------------------------
  for (int t = 0; t < T_; ++t) {
    k_pack_x<<<(B_*64 + 255)/256, 256, 0, stream>>>(s_f32, act_seq, t, x_bf);
    gemm(0, x_bf, 64,   Wb[W_GIH], 64,   Bb[W_GIH], gx_f, 600, 600, nullptr,0, B_, 608, 64 );
    gemm(0, h_bf, 224,  Wb[W_GHH], 224,  Bb[W_GHH], gh_f, 600, 600, nullptr,0, B_, 608, 224);
    k_gate<<<(B_*224 + 255)/256, 256, 0, stream>>>(gx_f, gh_f, h_f32, h_bf, sc_bf, t);
    k_pack_xpos<<<(B_*1248 + 255)/256, 256, 0, stream>>>(h_f32, emb_bf, t, xpos_bf);
    gemm(1, xpos_bf, 1248, Wb[W_POS1], 1248, Bb[W_POS1], nullptr,0,H_,  p1_bf, 224, B_, 224, 1248);
    gemm(0, p1_bf,   224,  Wb[W_POS2], 224,  Bb[W_POS2], post_f, 2*S_, 2*S_, nullptr,0, B_, 64, 224);
    gemm(1, h_bf,    224,  Wb[W_PRI1], 224,  Bb[W_PRI1], nullptr,0,H_,  q1_bf, 224, B_, 224, 224 );
    gemm(0, q1_bf,   224,  Wb[W_PRI2], 224,  Bb[W_PRI2], pri_f,  2*S_, 2*S_, nullptr,0, B_, 64, 224);
    k_skl<<<(B_ + 255)/256, 256, 0, stream>>>(post_f, pri_f, noise, t, s_f32, sc_bf, kl_out);
  }

  // ---- decoder: sc(256) -> 512 -> 256 -> OBS ------------------------------
  gemm(1, sc_bf, 256, Wb[W_DEC1], 256, Bb[W_DEC1], nullptr,0,512,  bufA, 512, BT_, 512, 256);
  gemm(1, bufA,  512, Wb[W_DEC2], 512, Bb[W_DEC2], nullptr,0,256,  bufB, 256, BT_, 256, 512);
  gemm(0, bufB,  256, Wb[W_DEC3], 256, Bb[W_DEC3], recon_out, OBS_, OBS_, nullptr,0, BT_, 64, 256);

  // ---- reward head: sc -> 256 -> 128 -> 1 ---------------------------------
  gemm(1, sc_bf, 256, Wb[W_REW1], 256, Bb[W_REW1], nullptr,0,256,  bufB, 256, BT_, 256, 256);
  gemm(1, bufB,  256, Wb[W_REW2], 256, Bb[W_REW2], nullptr,0,128,  bufA, 128, BT_, 128, 256);
  gemm(0, bufA,  128, Wb[W_REW3], 128, Bb[W_REW3], rew_out, 1, 1,  nullptr,0, BT_, 16, 128);

  // ---- continue head: sc -> 128 -> 64 -> 1 (sigmoid) ----------------------
  gemm(1, sc_bf, 256, Wb[W_CON1], 256, Bb[W_CON1], nullptr,0,128,  bufB, 128, BT_, 128, 256);
  gemm(1, bufB,  128, Wb[W_CON2], 128, Bb[W_CON2], nullptr,0,64,   bufA, 64,  BT_, 64,  128);
  gemm(2, bufA,  64,  Wb[W_CON3], 64,  Bb[W_CON3], con_out, 1, 1,  nullptr,0, BT_, 16, 64 );
}
